// sda_88742614270714
// MI455X (gfx1250) — matrix operand order repaired and checked by reading, not compiled, not run on hardware
//
#include <hip/hip_runtime.h>
#include <hip/hip_bf16.h>

typedef __attribute__((ext_vector_type(16))) _Float16 v16h;
typedef __attribute__((ext_vector_type(8)))  float    v8f;

#define ATT_B 2
#define ATT_H 12
#define ATT_S 2048
#define ATT_D 64
#define QT    16
#define WS_LD 2052                                  // padded f32 row stride (multiple of 4)
#define WS_FLOATS (QT * WS_LD)                      // 32832 floats = 131328 B
#define PACC_FLOATS (8 * 16 * 16)                   // 2048 floats = 8192 B
#define SHMEM_BYTES ((size_t)(WS_FLOATS + PACC_FLOATS) * sizeof(float))

// pack a float4 into 4 consecutive f16 elements of a v16h
#define CVT4(dst, base, f4)                         \
    do {                                            \
        (dst)[(base) + 0] = (_Float16)(f4).x;       \
        (dst)[(base) + 1] = (_Float16)(f4).y;       \
        (dst)[(base) + 2] = (_Float16)(f4).z;       \
        (dst)[(base) + 3] = (_Float16)(f4).w;       \
    } while (0)

__global__ __launch_bounds__(256, 2)
void attn_fwd_kernel(const float* __restrict__ q,
                     const float* __restrict__ k,
                     const float* __restrict__ v,
                     const float* __restrict__ mask,
                     float* __restrict__ out,
                     float* __restrict__ we)
{
    extern __shared__ char smem[];
    float* ws   = (float*)smem;                         // [QT][WS_LD] logits -> probs
    float* pacc = (float*)(smem + (size_t)WS_FLOATS*4); // [8][16][16] phase-2 partials

    const int tid  = threadIdx.x;
    const int wave = tid >> 5;
    const int lane = tid & 31;

    const int bh = blockIdx.x >> 7;            // 128 query tiles per (b,h)
    const int qt = blockIdx.x & 127;
    const int q0 = qt * QT;

    const size_t bh_off = (size_t)bh * ATT_S * ATT_D;
    const float* qbase = q + bh_off + (size_t)q0 * ATT_D;
    const float* kbase = k + bh_off;
    const float* vbase = v + bh_off;

    const int n_  = lane & 15;                 // N (and A-row) index for this lane
    const int mlo = (lane < 16) ? 0 : 8;       // C/D row base (f32 16x16 layout)
    const int alo = (lane < 16) ? 0 : 16;      // A-matrix K-chunk base (16-bit 16x32)
    const int blo = (lane < 16) ? 0 : 16;      // B-matrix K-chunk base (16-bit 32x16)

    // ---- Q A-fragments (K = 0..31 and 32..63), contiguous k = 16h + s per lane half --------
    v16h qa0, qa1;
    {
        const float* p = qbase + (size_t)n_ * ATT_D;   // A row m = lane&15
        const float4* p4 = (const float4*)p;           // 16B-aligned chunks
        float4 c0 = p4[(alo)      >> 2], c1 = p4[(alo + 4)  >> 2];
        float4 c2 = p4[(alo + 8)  >> 2], c3 = p4[(alo + 12) >> 2];
        float4 c4 = p4[(alo + 32) >> 2], c5 = p4[(alo + 36) >> 2];
        float4 c6 = p4[(alo + 40) >> 2], c7 = p4[(alo + 44) >> 2];
        CVT4(qa0, 0, c0);  CVT4(qa0, 4, c1);
        CVT4(qa0, 8, c2);  CVT4(qa0, 12, c3);
        CVT4(qa1, 0, c4);  CVT4(qa1, 4, c5);
        CVT4(qa1, 8, c6);  CVT4(qa1, 12, c7);
    }

    // ================= Phase 1: scaled logits = Q K^T * 1/sqrt(D) =================
    #pragma unroll 1
    for (int t = 0; t < 16; ++t) {
        const int j = wave + (t << 3);                     // key tile 0..127
        const float* kt = kbase + (size_t)j * 16 * ATT_D;
        if (t + 1 < 16)
            __builtin_prefetch(kbase + (size_t)(j + 8) * 16 * ATT_D, 0, 3);

        // B[d][n] = K[n][d]: lane owns key n_, two contiguous 16-float chunks
        v16h b0, b1;
        {
            const float4* kp4 = (const float4*)(kt + (size_t)n_ * ATT_D);
            float4 c0 = kp4[(blo)      >> 2], c1 = kp4[(blo + 4)  >> 2];
            float4 c2 = kp4[(blo + 8)  >> 2], c3 = kp4[(blo + 12) >> 2];
            float4 c4 = kp4[(blo + 32) >> 2], c5 = kp4[(blo + 36) >> 2];
            float4 c6 = kp4[(blo + 40) >> 2], c7 = kp4[(blo + 44) >> 2];
            CVT4(b0, 0, c0);  CVT4(b0, 4, c1);
            CVT4(b0, 8, c2);  CVT4(b0, 12, c3);
            CVT4(b1, 0, c4);  CVT4(b1, 4, c5);
            CVT4(b1, 8, c6);  CVT4(b1, 12, c7);
        }

        v8f acc = {};
        acc = __builtin_amdgcn_wmma_f32_16x16x32_f16(false, qa0, false, b0,
                                                     (short)0, acc, false, false);
        acc = __builtin_amdgcn_wmma_f32_16x16x32_f16(false, qa1, false, b1,
                                                     (short)0, acc, false, false);

        const int col = (j << 4) + n_;
        #pragma unroll
        for (int e = 0; e < 8; ++e)
            ws[(mlo + e) * WS_LD + col] = acc[e] * 0.125f;
    }
    __syncthreads();

    // ==== Softmax (2 rows per wave): mask add fused into the coalesced max pass ===
    const size_t we_base = (size_t)bh * ATT_S * ATT_S;
    #pragma unroll
    for (int rr = 0; rr < 2; ++rr) {
        const int r = (wave << 1) + rr;
        float4* row4 = (float4*)(ws + r * WS_LD);          // 16B aligned (WS_LD%4==0)
        const float4* mrow4 = (const float4*)(mask + (size_t)(q0 + r) * ATT_S);

        float mx = -3.0e38f;
        for (int i = lane; i < ATT_S / 4; i += 32) {       // b128 mask stream + add
            float4 x = row4[i];
            float4 mk = mrow4[i];
            x.x += mk.x; x.y += mk.y; x.z += mk.z; x.w += mk.w;
            row4[i] = x;
            mx = fmaxf(mx, fmaxf(fmaxf(x.x, x.y), fmaxf(x.z, x.w)));
        }
        #pragma unroll
        for (int o = 16; o; o >>= 1) mx = fmaxf(mx, __shfl_xor(mx, o, 32));

        float sum = 0.f;
        for (int i = lane; i < ATT_S / 4; i += 32) {
            float4 x = row4[i];
            x.x = __expf(x.x - mx); x.y = __expf(x.y - mx);
            x.z = __expf(x.z - mx); x.w = __expf(x.w - mx);
            row4[i] = x;
            sum += (x.x + x.y) + (x.z + x.w);
        }
        #pragma unroll
        for (int o = 16; o; o >>= 1) sum += __shfl_xor(sum, o, 32);
        const float inv = 1.0f / sum;

        float4* wrow4 = (float4*)(we + we_base + (size_t)(q0 + r) * ATT_S);
        for (int i = lane; i < ATT_S / 4; i += 32) {       // coalesced b128 stream
            float4 p = row4[i];
            p.x *= inv; p.y *= inv; p.z *= inv; p.w *= inv;
            row4[i]  = p;
            wrow4[i] = p;
        }
    }
    __syncthreads();

    // ================= Phase 2: out = P V =========================================
    const int ntile = wave & 3;                 // output dim tile (4 x 16 = 64)
    const int ks0   = (wave >> 2) << 5;         // K-step half: 0 or 32 (32 keys/step)
    const int nd    = (ntile << 4) + n_;

    v8f oacc = {};
    #pragma unroll 1
    for (int kt2 = 0; kt2 < 32; ++kt2) {
        const int kb = (ks0 + kt2) << 5;        // first key of this 32-key step

        v16h pa;                                 // P tile as f16 A-fragment (LDS b128)
        {
            const float4* pr4 = (const float4*)(ws + n_ * WS_LD + kb);
            float4 c0 = pr4[(alo)      >> 2], c1 = pr4[(alo + 4)  >> 2];
            float4 c2 = pr4[(alo + 8)  >> 2], c3 = pr4[(alo + 12) >> 2];
            CVT4(pa, 0, c0);  CVT4(pa, 4, c1);
            CVT4(pa, 8, c2);  CVT4(pa, 12, c3);
        }

        v16h vb;                                 // V as B-fragment: B[k][n] = V[key][n]
        const float* vp = vbase + (size_t)(kb + blo) * ATT_D + nd;
        #pragma unroll
        for (int i = 0; i < 16; ++i) vb[i] = (_Float16)vp[i * ATT_D];

        oacc = __builtin_amdgcn_wmma_f32_16x16x32_f16(false, pa, false, vb,
                                                      (short)0, oacc, false, false);
    }

    {   // stash partials for cross-wave reduction
        float* pw = pacc + wave * 256;
        #pragma unroll
        for (int e = 0; e < 8; ++e) pw[(mlo + e) * 16 + n_] = oacc[e];
    }
    __syncthreads();

    if (wave < 4) {                              // ntile == wave here
        const float* p0 = pacc + wave * 256;
        const float* p1 = pacc + (wave + 4) * 256;
        #pragma unroll
        for (int e = 0; e < 8; ++e) {
            const int m = mlo + e;
            out[bh_off + (size_t)(q0 + m) * ATT_D + (wave << 4) + n_] =
                p0[m * 16 + n_] + p1[m * 16 + n_];
        }
    }
}

extern "C" void kernel_launch(void* const* d_in, const int* in_sizes, int n_in,
                              void* d_out, int out_size, void* d_ws, size_t ws_size,
                              hipStream_t stream) {
    (void)in_sizes; (void)n_in; (void)out_size; (void)d_ws; (void)ws_size;
    const float* q    = (const float*)d_in[0];
    const float* k    = (const float*)d_in[1];
    const float* v    = (const float*)d_in[2];
    const float* mask = (const float*)d_in[3];
    float* out = (float*)d_out;
    float* we  = out + (size_t)ATT_B * ATT_H * ATT_S * ATT_D;

    static_assert(SHMEM_BYTES <= 160 * 1024, "2 workgroups per 320KB WGP");
    hipFuncSetAttribute((const void*)attn_fwd_kernel,
                        hipFuncAttributeMaxDynamicSharedMemorySize,
                        (int)SHMEM_BYTES);

    dim3 grid(ATT_B * ATT_H * (ATT_S / QT));   // 3072 workgroups
    dim3 block(256);                            // 8 wave32s
    attn_fwd_kernel<<<grid, block, SHMEM_BYTES, stream>>>(q, k, v, mask, out, we);
}
